// GraphTransformer_3659312136586
// MI455X (gfx1250) — matrix operand order repaired and checked by reading, not compiled, not run on hardware
//
#include <hip/hip_runtime.h>
#include <hip/hip_bf16.h>

#define N_NODES  16384
#define N_GRAPHS 64
#define NPG      256
#define F_IN     128
#define E_DIM    256
#define N_HEADSC 8
#define MLP_DIM  1024
#define N_LAYERSC 4
#define N_CLASSES 10
#define N_EDGESC 262144
#define S_PAD    288            // 257 padded to multiple of 32
#define S_REAL   257
#define LN_EPSF  1e-5f

typedef __bf16 bf16_t;
typedef __bf16 v16bf __attribute__((ext_vector_type(16)));
typedef __bf16 v8bf  __attribute__((ext_vector_type(8)));
typedef float  v8f   __attribute__((ext_vector_type(8)));

// ---------------------------------------------------------------- WMMA helpers
__device__ inline v8f wmma_bf16(v16bf a, v16bf b, v8f c) {
  // D = A(16x32) * B(32x16) + C, fp32 accumulate
  return __builtin_amdgcn_wmma_f32_16x16x32_bf16(false, a, false, b, (short)0, c, false, false);
}

// A operand, 16x32 (MxK), source already bf16 (global or LDS):
// lane L: m = L&15, kb = (L>>4)*16 ; holds K = kb..kb+15
__device__ inline v16bf load_a_bf16(const bf16_t* A, int lda, int mBase, int kBase) {
  int lane = threadIdx.x & 31;
  int m  = mBase + (lane & 15);
  int kb = (lane >> 4) << 4;
  const bf16_t* r = A + (size_t)m * lda + kBase + kb;
  v8bf lo = *(const v8bf*)r;          // 16B packed load
  v8bf hi = *(const v8bf*)(r + 8);    // 16B packed load
  v16bf a;
#pragma unroll
  for (int i = 0; i < 8; ++i) { a[i] = lo[i]; a[8 + i] = hi[i]; }
  return a;
}

// A operand from fp32 source (attention Q path only)
__device__ inline v16bf load_a_f32(const float* A, int lda, int mBase, int kBase) {
  int lane = threadIdx.x & 31;
  int m  = mBase + (lane & 15);
  int kb = (lane >> 4) << 4;
  const float* r = A + (size_t)m * lda + kBase + kb;
  const float4* p0 = (const float4*)r;
  const float4* p1 = (const float4*)(r + 8);
  float4 x0 = p0[0], x1 = p0[1], y0 = p1[0], y1 = p1[1];
  v16bf a;
  a[0]=(__bf16)x0.x; a[1]=(__bf16)x0.y; a[2]=(__bf16)x0.z; a[3]=(__bf16)x0.w;
  a[4]=(__bf16)x1.x; a[5]=(__bf16)x1.y; a[6]=(__bf16)x1.z; a[7]=(__bf16)x1.w;
  a[8]=(__bf16)y0.x; a[9]=(__bf16)y0.y; a[10]=(__bf16)y0.z; a[11]=(__bf16)y0.w;
  a[12]=(__bf16)y1.x; a[13]=(__bf16)y1.y; a[14]=(__bf16)y1.z; a[15]=(__bf16)y1.w;
  return a;
}

// B operand from bf16 W stored [N][K] row-major, computing A*W^T:
// B[k][n] = W[n*ldw + kBase + k]; lane L: n = L&15 (clamped), kb=(L>>4)*16;
// 16 contiguous bf16 per lane -> one 32B vector load.
__device__ inline v16bf load_bT_bf16(const bf16_t* W, int ldw, int nBase, int kBase, int Nlim) {
  int lane = threadIdx.x & 31;
  int n  = nBase + (lane & 15);
  n = (n < Nlim) ? n : (Nlim - 1);
  int kb = (lane >> 4) << 4;
  return *(const v16bf*)(W + (size_t)n * ldw + kBase + kb);
}

// B operand from row-major bf16 B[K][N] (strided; attention V path)
__device__ inline v16bf load_b_bf16(const bf16_t* B, int ldb, int kBase, int nBase) {
  int lane = threadIdx.x & 31;
  int n  = nBase + (lane & 15);
  int kb = ((lane >> 4) << 4) + kBase;
  const bf16_t* p = B + (size_t)kb * ldb + n;
  v16bf b;
#pragma unroll
  for (int v = 0; v < 8; ++v) {
    b[2*v]   = p[(2*v) * ldb];
    b[2*v+1] = p[(2*v+1) * ldb];
  }
  return b;
}

// ----------------------------------------------------------- fp32 -> bf16 pass
__global__ void k_cvt(const float* __restrict__ s, bf16_t* __restrict__ d, int n) {
  int t = blockIdx.x * blockDim.x + threadIdx.x;
  int i = t * 4;
  if (i + 3 < n) {
    float4 x = *(const float4*)(s + i);
    d[i+0] = (bf16_t)x.x; d[i+1] = (bf16_t)x.y; d[i+2] = (bf16_t)x.z; d[i+3] = (bf16_t)x.w;
  } else {
    for (; i < n; ++i) d[i] = (bf16_t)s[i];
  }
}

// ------------------------------------------------------------ generic WMMA GEMM
// C[M x N] = act(A_bf16[M x K] @ W_bf16[N x K]^T + bias [+ add])
// 1 wave computes a 32x64 macro-tile (2 M-tiles x 4 N-tiles, 8 accumulators);
// block = 4 waves = 32 x 256 of C.  Requires M % 32 == 0, K % 32 == 0.
__global__ __launch_bounds__(128) void k_gemm(
    const bf16_t* __restrict__ A, int lda,
    const bf16_t* __restrict__ W, int ldw,
    const float* __restrict__ bias,
    const float* __restrict__ add,
    float* __restrict__ C, int ldc,
    int M, int N, int K, int relu)
{
  int wave  = threadIdx.x >> 5;
  int nBase = (blockIdx.x * 4 + wave) * 64;
  int mBase = blockIdx.y * 32;
  if (nBase >= N) return;               // wave-uniform exit

  v8f acc[2][4];
#pragma unroll
  for (int h = 0; h < 2; ++h)
#pragma unroll
    for (int j = 0; j < 4; ++j) acc[h][j] = (v8f){};

  for (int k = 0; k < K; k += 32) {
    v16bf a0 = load_a_bf16(A, lda, mBase,      k);
    v16bf a1 = load_a_bf16(A, lda, mBase + 16, k);
#pragma unroll
    for (int j = 0; j < 4; ++j) {
      v16bf b = load_bT_bf16(W, ldw, nBase + j * 16, k, N);
      acc[0][j] = wmma_bf16(a0, b, acc[0][j]);
      acc[1][j] = wmma_bf16(a1, b, acc[1][j]);
    }
  }

  int lane = threadIdx.x & 31;
  int nl = lane & 15;
  int mo = (lane >> 4) << 3;
#pragma unroll
  for (int j = 0; j < 4; ++j) {
    int n = nBase + j * 16 + nl;
    if (n < N) {
      float bv = bias ? bias[n] : 0.0f;
#pragma unroll
      for (int h = 0; h < 2; ++h) {
#pragma unroll
        for (int i = 0; i < 8; ++i) {
          int m = mBase + h * 16 + mo + i;
          float v = acc[h][j][i] + bv;
          if (add)  v += add[(size_t)m * ldc + n];
          if (relu) v = fmaxf(v, 0.0f);
          C[(size_t)m * ldc + n] = v;
        }
      }
    }
  }
}

// ------------------------------------------------------------------ GCN kernels
__global__ void k_fill(float* p, float v, int n) {
  int i = blockIdx.x * blockDim.x + threadIdx.x;
  if (i < n) p[i] = v;
}
__global__ void k_deg(const int* __restrict__ col, float* __restrict__ deg, int nE) {
  int e = blockIdx.x * blockDim.x + threadIdx.x;
  if (e < nE) atomicAdd(&deg[col[e]], 1.0f);
}
__global__ void k_dinv(const float* __restrict__ deg, float* __restrict__ dinv, int n) {
  int i = blockIdx.x * blockDim.x + threadIdx.x;
  if (i < n) dinv[i] = rsqrtf(deg[i]);
}
__global__ void k_self(const float* __restrict__ xw, const float* __restrict__ dinv,
                       float* __restrict__ out, int nNodes) {
  int t = blockIdx.x * blockDim.x + threadIdx.x;
  if (t < nNodes * F_IN) {
    int i = t >> 7;
    float di = dinv[i];
    out[t] = di * di * xw[t];
  }
}
__global__ void k_scatter(const int* __restrict__ row, const int* __restrict__ col,
                          const float* __restrict__ xw, const float* __restrict__ dinv,
                          float* __restrict__ out, int nE) {
  int t = blockIdx.x * blockDim.x + threadIdx.x;   // nE * 32 threads, 4 floats each
  int e  = t >> 5;
  int f4 = (t & 31) << 2;
  if (e < nE) {
    int r = row[e], c = col[e];
    float nrm = dinv[r] * dinv[c];
    const float4 s = *(const float4*)(xw + (size_t)r * F_IN + f4);
    float* dst = out + (size_t)c * F_IN + f4;
    atomicAdd(dst + 0, nrm * s.x);
    atomicAdd(dst + 1, nrm * s.y);
    atomicAdd(dst + 2, nrm * s.z);
    atomicAdd(dst + 3, nrm * s.w);
  }
}
__global__ void k_bias_relu(float* __restrict__ o, const float* __restrict__ b, int n) {
  int t = blockIdx.x * blockDim.x + threadIdx.x;
  if (t < n) o[t] = fmaxf(o[t] + b[t & (F_IN - 1)], 0.0f);
}

// ----------------------------------------------------- assemble padded H tensor
__global__ void k_build_h(const float* __restrict__ emb, const float* __restrict__ cls,
                          float* __restrict__ H, int total) {
  int t = blockIdx.x * blockDim.x + threadIdx.x;
  if (t < total) {
    int c  = t & (E_DIM - 1);
    int gs = t >> 8;
    int s  = gs % S_PAD;
    int g  = gs / S_PAD;
    float v;
    if (s == 0)            v = cls[c];
    else if (s <= NPG)     v = emb[((size_t)g * NPG + (s - 1)) * E_DIM + c];
    else                   v = 0.0f;
    H[t] = v;
  }
}

// ------------------------------------------------------------- attention kernel
// one block per (graph, head); 4 waves; qkv layout [G][S_PAD][768]
// K/V/P all staged in LDS as bf16 (WMMA native), ~74 KB per block.
__global__ __launch_bounds__(128) void k_attn(const float* __restrict__ qkv,
                                              float* __restrict__ att) {
  __shared__ __align__(32) bf16_t Ksh[S_PAD * 32];
  __shared__ __align__(32) bf16_t Vsh[S_PAD * 32];
  __shared__ __align__(32) bf16_t Psh[4 * 16 * S_PAD];

  int g  = blockIdx.x >> 3;
  int hh = blockIdx.x & 7;
  const float* base = qkv + (size_t)g * S_PAD * (3 * E_DIM);
  const float* Q  = base + hh * 32;
  const float* Kp = base + E_DIM + hh * 32;
  const float* Vp = base + 2 * E_DIM + hh * 32;

  for (int i = threadIdx.x; i < S_PAD * 32; i += 128) {
    int s = i >> 5, d = i & 31;
    Ksh[i] = (bf16_t)Kp[(size_t)s * (3 * E_DIM) + d];
    Vsh[i] = (bf16_t)Vp[(size_t)s * (3 * E_DIM) + d];
  }
  __syncthreads();

  int wave = threadIdx.x >> 5;
  int lane = threadIdx.x & 31;
  int ncol = lane & 15;
  int mo   = (lane >> 4) << 3;
  const float scale = 0.17677669529663687f;   // 1/sqrt(32)
  bf16_t* P = Psh + wave * 16 * S_PAD;

  for (int qt = wave; qt < S_PAD / 16; qt += 4) {
    // ---- scores S = Q Kt  (dh=32 == one WMMA K step)
    v16bf aq = load_a_f32(Q, 3 * E_DIM, qt * 16, 0);
    float sc[18][8];
#pragma unroll
    for (int j = 0; j < 18; ++j) {
      v8f acc = (v8f){};
      v16bf bk = load_bT_bf16(Ksh, 32, j * 16, 0, 1 << 30);   // B[d][n] = K[n][d]
      acc = wmma_bf16(aq, bk, acc);
#pragma unroll
      for (int i = 0; i < 8; ++i) sc[j][i] = acc[i];
    }
    // ---- masked softmax per row (row m = i + mo ; col = j*16 + ncol)
#pragma unroll
    for (int i = 0; i < 8; ++i) {
      float mx = -1e30f;
#pragma unroll
      for (int j = 0; j < 18; ++j) {
        float v = sc[j][i] * scale;
        sc[j][i] = v;
        if (j * 16 + ncol < S_REAL) mx = fmaxf(mx, v);
      }
#pragma unroll
      for (int off = 1; off < 16; off <<= 1) mx = fmaxf(mx, __shfl_xor(mx, off, 32));
      float sum = 0.0f;
#pragma unroll
      for (int j = 0; j < 18; ++j) {
        float e = (j * 16 + ncol < S_REAL) ? __expf(sc[j][i] - mx) : 0.0f;
        sc[j][i] = e;
        sum += e;
      }
#pragma unroll
      for (int off = 1; off < 16; off <<= 1) sum += __shfl_xor(sum, off, 32);
      float inv = 1.0f / sum;
#pragma unroll
      for (int j = 0; j < 18; ++j) sc[j][i] *= inv;
    }
    // ---- spill P as bf16 (D-layout regs -> row-major LDS) for A-operand reload
#pragma unroll
    for (int i = 0; i < 8; ++i) {
      int m = mo + i;
#pragma unroll
      for (int j = 0; j < 18; ++j)
        P[m * S_PAD + j * 16 + ncol] = (bf16_t)sc[j][i];
    }
    // ---- O = P V   (K stepped by 32, N split in two 16-halves of dh=32)
    v8f o0 = (v8f){}, o1 = (v8f){};
    for (int kc = 0; kc < S_PAD / 32; ++kc) {
      v16bf ap = load_a_bf16(P, S_PAD, 0, kc * 32);
      v16bf b0 = load_b_bf16(Vsh, 32, kc * 32, 0);
      v16bf b1 = load_b_bf16(Vsh, 32, kc * 32, 16);
      o0 = wmma_bf16(ap, b0, o0);
      o1 = wmma_bf16(ap, b1, o1);
    }
    float* outp = att + ((size_t)g * S_PAD + qt * 16) * E_DIM + hh * 32;
#pragma unroll
    for (int i = 0; i < 8; ++i) {
      int m = mo + i;
      outp[(size_t)m * E_DIM + ncol]      = o0[i];
      outp[(size_t)m * E_DIM + 16 + ncol] = o1[i];
    }
  }
}

// ------------------------------------------------------- residual add + LayerNorm
__global__ void k_add_ln(float* __restrict__ H, const float* __restrict__ R,
                         const float* __restrict__ g, const float* __restrict__ b,
                         int rows) {
  int row = blockIdx.x * (blockDim.x >> 5) + (threadIdx.x >> 5);
  if (row >= rows) return;
  int lane = threadIdx.x & 31;
  float* hr = H + (size_t)row * E_DIM;
  const float* rr = R + (size_t)row * E_DIM;
  float v[8];
  float s = 0.0f;
#pragma unroll
  for (int i = 0; i < 8; ++i) { v[i] = hr[lane + 32 * i] + rr[lane + 32 * i]; s += v[i]; }
#pragma unroll
  for (int off = 16; off; off >>= 1) s += __shfl_xor(s, off, 32);
  float mu = s * (1.0f / E_DIM);
  float var = 0.0f;
#pragma unroll
  for (int i = 0; i < 8; ++i) { float d = v[i] - mu; var += d * d; }
#pragma unroll
  for (int off = 16; off; off >>= 1) var += __shfl_xor(var, off, 32);
  float inv = rsqrtf(var * (1.0f / E_DIM) + LN_EPSF);
#pragma unroll
  for (int i = 0; i < 8; ++i)
    hr[lane + 32 * i] = (v[i] - mu) * inv * g[lane + 32 * i] + b[lane + 32 * i];
}

__global__ void k_gather_cls(const float* __restrict__ H, float* __restrict__ c, int n) {
  int t = blockIdx.x * blockDim.x + threadIdx.x;
  if (t < n) {
    int g = t >> 8, ch = t & (E_DIM - 1);
    c[t] = H[(size_t)g * S_PAD * E_DIM + ch];
  }
}

// --------------------------------------------------------------------- launcher
static inline void cvt_async(const float* s, bf16_t* d, int n, hipStream_t st) {
  k_cvt<<<(n / 4 + 255) / 256 + 1, 256, 0, st>>>(s, d, n);
}

extern "C" void kernel_launch(void* const* d_in, const int* in_sizes, int n_in,
                              void* d_out, int out_size, void* d_ws, size_t ws_size,
                              hipStream_t stream) {
  const float* x     = (const float*)d_in[0];
  const int*   ei    = (const int*)  d_in[1];
  const float* pos   = (const float*)d_in[3];
  const float* gcn_w = (const float*)d_in[4];
  const float* gcn_b = (const float*)d_in[5];
  const float* clsv  = (const float*)d_in[6];
  const float* emb_w = (const float*)d_in[7];
  const float* emb_b = (const float*)d_in[8];
  const float* in_w  = (const float*)d_in[9];
  const float* in_b  = (const float*)d_in[10];
  const float* out_w = (const float*)d_in[11];
  const float* out_b = (const float*)d_in[12];
  const float* l1w   = (const float*)d_in[13];
  const float* l1b   = (const float*)d_in[14];
  const float* l2w   = (const float*)d_in[15];
  const float* l2b   = (const float*)d_in[16];
  const float* g1    = (const float*)d_in[17];
  const float* b1    = (const float*)d_in[18];
  const float* g2    = (const float*)d_in[19];
  const float* b2    = (const float*)d_in[20];
  const float* fcw   = (const float*)d_in[21];
  const float* fcb   = (const float*)d_in[22];
  float* out = (float*)d_out;
  float* ws  = (float*)d_ws;

  const size_t FH = (size_t)N_GRAPHS * S_PAD * E_DIM;          // 4,718,592 floats
  float* H    = ws;
  float* emb  = ws + FH;
  float* xw   = emb + (size_t)N_NODES * E_DIM;
  float* gcn  = xw  + (size_t)N_NODES * F_IN;
  float* deg  = gcn + (size_t)N_NODES * F_IN;
  float* dinv = deg + N_NODES;
  float* qkv  = ws + FH;                                        // reuse (emb..dinv dead)
  float* attb = qkv + (size_t)N_GRAPHS * S_PAD * 3 * E_DIM;
  float* tmp  = attb + FH;
  float* fbuf = ws + FH;                                        // reuse qkv/att region
  float* clsb = tmp + FH;

  // bf16 staging area (weights converted once + activation scratch per GEMM)
  bf16_t* gcnWb = (bf16_t*)(clsb + (size_t)N_GRAPHS * E_DIM);
  bf16_t* embWb = gcnWb + F_IN * F_IN;                          //  16384
  bf16_t* inWb  = embWb + E_DIM * F_IN;                         //  32768
  bf16_t* outWb = inWb  + (size_t)N_LAYERSC * 3 * E_DIM * E_DIM; // 786432
  bf16_t* l1Wb  = outWb + (size_t)N_LAYERSC * E_DIM * E_DIM;    // 262144
  bf16_t* l2Wb  = l1Wb  + (size_t)N_LAYERSC * MLP_DIM * E_DIM;  // 1048576
  bf16_t* fcWb  = l2Wb  + (size_t)N_LAYERSC * E_DIM * MLP_DIM;  // 1048576
  bf16_t* abuf  = fcWb  + N_CLASSES * E_DIM;                    // act scratch (<= 18.9M)

  const int Mrows = N_GRAPHS * S_PAD;                           // 18432

  // ---- weights -> bf16 (once)
  cvt_async(gcn_w, gcnWb, F_IN * F_IN, stream);
  cvt_async(emb_w, embWb, E_DIM * F_IN, stream);
  cvt_async(in_w,  inWb,  N_LAYERSC * 3 * E_DIM * E_DIM, stream);
  cvt_async(out_w, outWb, N_LAYERSC * E_DIM * E_DIM, stream);
  cvt_async(l1w,   l1Wb,  N_LAYERSC * MLP_DIM * E_DIM, stream);
  cvt_async(l2w,   l2Wb,  N_LAYERSC * E_DIM * MLP_DIM, stream);
  cvt_async(fcw,   fcWb,  N_CLASSES * E_DIM, stream);

  // ---- GCN conv:  xw = x @ gcn_w^T
  cvt_async(x, abuf, N_NODES * F_IN, stream);
  k_gemm<<<dim3(1, N_NODES / 32), 128, 0, stream>>>(
      abuf, F_IN, gcnWb, F_IN, nullptr, nullptr, xw, F_IN, N_NODES, F_IN, F_IN, 0);
  k_fill<<<(N_NODES + 255) / 256, 256, 0, stream>>>(deg, 1.0f, N_NODES);
  k_deg<<<(N_EDGESC + 255) / 256, 256, 0, stream>>>(ei + N_EDGESC, deg, N_EDGESC);
  k_dinv<<<(N_NODES + 255) / 256, 256, 0, stream>>>(deg, dinv, N_NODES);
  k_self<<<(N_NODES * F_IN + 255) / 256, 256, 0, stream>>>(xw, dinv, gcn, N_NODES);
  k_scatter<<<(N_EDGESC * 32 + 255) / 256, 256, 0, stream>>>(ei, ei + N_EDGESC, xw, dinv, gcn, N_EDGESC);
  k_bias_relu<<<(N_NODES * F_IN + 255) / 256, 256, 0, stream>>>(gcn, gcn_b, N_NODES * F_IN);

  // ---- embed + pos_enc
  cvt_async(gcn, abuf, N_NODES * F_IN, stream);
  k_gemm<<<dim3(1, N_NODES / 32), 128, 0, stream>>>(
      abuf, F_IN, embWb, F_IN, emb_b, pos, emb, E_DIM, N_NODES, E_DIM, F_IN, 0);

  // ---- build padded H with CLS token
  int nH = (int)FH;
  k_build_h<<<(nH + 255) / 256, 256, 0, stream>>>(emb, clsv, H, nH);

  // ---- transformer layers
  for (int l = 0; l < N_LAYERSC; ++l) {
    cvt_async(H, abuf, Mrows * E_DIM, stream);
    k_gemm<<<dim3(3, Mrows / 32), 128, 0, stream>>>(
        abuf, E_DIM, inWb + (size_t)l * 3 * E_DIM * E_DIM, E_DIM,
        in_b + (size_t)l * 3 * E_DIM, nullptr, qkv, 3 * E_DIM, Mrows, 3 * E_DIM, E_DIM, 0);
    k_attn<<<N_GRAPHS * N_HEADSC, 128, 0, stream>>>(qkv, attb);
    cvt_async(attb, abuf, Mrows * E_DIM, stream);
    k_gemm<<<dim3(1, Mrows / 32), 128, 0, stream>>>(
        abuf, E_DIM, outWb + (size_t)l * E_DIM * E_DIM, E_DIM,
        out_b + (size_t)l * E_DIM, nullptr, tmp, E_DIM, Mrows, E_DIM, E_DIM, 0);
    k_add_ln<<<(Mrows + 7) / 8, 256, 0, stream>>>(H, tmp, g1 + (size_t)l * E_DIM, b1 + (size_t)l * E_DIM, Mrows);
    cvt_async(H, abuf, Mrows * E_DIM, stream);
    k_gemm<<<dim3(4, Mrows / 32), 128, 0, stream>>>(
        abuf, E_DIM, l1Wb + (size_t)l * MLP_DIM * E_DIM, E_DIM,
        l1b + (size_t)l * MLP_DIM, nullptr, fbuf, MLP_DIM, Mrows, MLP_DIM, E_DIM, 1);
    cvt_async(fbuf, abuf, Mrows * MLP_DIM, stream);
    k_gemm<<<dim3(1, Mrows / 32), 128, 0, stream>>>(
        abuf, MLP_DIM, l2Wb + (size_t)l * E_DIM * MLP_DIM, MLP_DIM,
        l2b + (size_t)l * E_DIM, nullptr, tmp, E_DIM, Mrows, E_DIM, MLP_DIM, 0);
    k_add_ln<<<(Mrows + 7) / 8, 256, 0, stream>>>(H, tmp, g2 + (size_t)l * E_DIM, b2 + (size_t)l * E_DIM, Mrows);
  }

  // ---- classifier head
  k_gather_cls<<<(N_GRAPHS * E_DIM + 255) / 256, 256, 0, stream>>>(H, clsb, N_GRAPHS * E_DIM);
  cvt_async(clsb, abuf, N_GRAPHS * E_DIM, stream);
  k_gemm<<<dim3(1, N_GRAPHS / 32), 128, 0, stream>>>(
      abuf, E_DIM, fcWb, E_DIM, fcb, nullptr, out, N_CLASSES, N_GRAPHS, N_CLASSES, E_DIM, 0);
}
